// SAModule_42159398978139
// MI455X (gfx1250) — matrix operand order repaired and checked by reading, not compiled, not run on hardware
//
#include <hip/hip_runtime.h>
#include <hip/hip_fp16.h>

typedef __attribute__((ext_vector_type(16))) _Float16 v16h;
typedef __attribute__((ext_vector_type(8)))  float    v8f;

#define DEVINL __device__ __forceinline__

static constexpr int BG    = 4;      // graphs
static constexpr int NPER  = 4096;   // points per graph
static constexpr int NPTS  = BG * NPER;
static constexpr int KNBR  = 64;     // neighbors
static constexpr int MFPS  = 1024;   // fps samples per graph
static constexpr int CIN   = 64;
static constexpr float R2  = 0.2f * 0.2f;

// d_ws byte offsets
static constexpr size_t WS_NBR  = 0;                                   // int32 [NPTS][64]
static constexpr size_t WS_FRAG = WS_NBR + (size_t)NPTS * 64 * 4;      // f16 frags 36*512
static constexpr size_t WS_OUT  = WS_FRAG + 36 * 512 * 2 + 256;        // f32 [NPTS][128]
static constexpr size_t WS_FPS  = WS_OUT + (size_t)NPTS * 128 * 4;     // int32 [BG][MFPS]

union V16H {
    v16h v;
    unsigned u[8];
    uint4 q[2];
    _Float16 h[16];
};

DEVINL unsigned pk2(float a, float b) {
    return __builtin_bit_cast(unsigned, __builtin_amdgcn_cvt_pkrtz(a, b));
}

// ---------------------------------------------------------------------------
// Kernel 0: pre-swizzle transposed weights into WMMA A-fragment order.
// A = W^T (M = out channel, K = in channel), 16x32 f16 fragments.
// A-frag element map: lane L (m=L%16, h=L/16), element e -> K = e + 16h
// frag ids: layer1 ot*3+kt (0..11), layer2 12+ot*2+kt, layer3 20+ot*2+kt
// ---------------------------------------------------------------------------
__global__ void prep_frags_kernel(const float* __restrict__ W1,
                                  const float* __restrict__ W2,
                                  const float* __restrict__ W3,
                                  _Float16* __restrict__ frags) {
    int idx = blockIdx.x * 256 + threadIdx.x;
    if (idx >= 36 * 512) return;
    int frag = idx >> 9;
    int r    = idx & 511;
    int lane = r >> 4;
    int e    = r & 15;
    int m    = lane & 15;
    int hh   = lane >> 4;
    int kmap = e + 16 * hh;

    const float* W; int Kin, Mout, ot, kt;
    if (frag < 12)       { W = W1; Kin = 67; Mout = 64;  ot = frag / 3;        kt = frag % 3; }
    else if (frag < 20)  { int f = frag - 12; W = W2; Kin = 64; Mout = 64;  ot = f / 2; kt = f % 2; }
    else                 { int f = frag - 20; W = W3; Kin = 64; Mout = 128; ot = f / 2; kt = f % 2; }

    int k  = kt * 32 + kmap;
    int mg = ot * 16 + m;
    float v = (k < Kin) ? W[(size_t)k * Mout + mg] : 0.0f;
    frags[(size_t)frag * 512 + lane * 16 + e] = (_Float16)v;
}

// ---------------------------------------------------------------------------
// Kernel 1: radius K-nearest per point (one block per point).
// Radix binary search on uint bits of d2 (inf outside ball) for 64th smallest.
// Invalid slots filled with self index (self edge is always selected anyway).
// ---------------------------------------------------------------------------
__global__ __launch_bounds__(256) void radius_kernel(const float* __restrict__ pos,
                                                     int* __restrict__ nbr) {
    __shared__ unsigned ebits[NPER];
    __shared__ unsigned s_cnt;
    __shared__ int s_emit;

    int i    = blockIdx.x;
    int base = (i >> 12) << 12;   // graph base
    int tid  = threadIdx.x;

    float pix = pos[3 * i], piy = pos[3 * i + 1], piz = pos[3 * i + 2];
    for (int t = tid; t < NPER; t += 256) {
        int j = base + t;
        float dx = pos[3 * j] - pix, dy = pos[3 * j + 1] - piy, dz = pos[3 * j + 2] - piz;
        float d2 = dx * dx + dy * dy + dz * dz;
        ebits[t] = (d2 <= R2) ? __float_as_uint(d2) : 0x7F800000u;
    }
    __syncthreads();

    unsigned lo = 0u, hi = 0x7F800000u;
    while (lo < hi) {
        unsigned mid = lo + ((hi - lo) >> 1);
        int c = 0;
        for (int t = tid; t < NPER; t += 256) c += (ebits[t] <= mid);
        #pragma unroll
        for (int off = 16; off; off >>= 1) c += __shfl_down(c, off, 32);
        if (tid == 0) s_cnt = 0;
        __syncthreads();
        if ((tid & 31) == 0) atomicAdd(&s_cnt, (unsigned)c);
        __syncthreads();
        unsigned total = s_cnt;
        __syncthreads();
        if (total >= 64u) hi = mid; else lo = mid + 1;
    }
    unsigned tstar = lo;

    if (tid == 0) s_emit = 0;
    __syncthreads();
    for (int t = tid; t < NPER; t += 256) {
        if (ebits[t] < tstar) {
            int p = atomicAdd(&s_emit, 1);        // guaranteed < 64
            nbr[(size_t)i * 64 + p] = base + t;
        }
    }
    __syncthreads();
    if (tstar != 0x7F800000u) {
        for (int t = tid; t < NPER; t += 256) {
            if (ebits[t] == tstar) {
                int p = atomicAdd(&s_emit, 1);
                if (p < 64) nbr[(size_t)i * 64 + p] = base + t;
            }
        }
    }
    __syncthreads();
    int c0 = s_emit; if (c0 > 64) c0 = 64;
    for (int p = c0 + tid; p < 64; p += 256) nbr[(size_t)i * 64 + p] = i;   // self-fill
}

// ---------------------------------------------------------------------------
// Kernel 2: PointConv MLP (67->64->64->128, relu) + max over 64 neighbors.
// One wave32 per point; computes H^T = W^T * F^T with v_wmma_f32_16x16x32_f16.
// D-tile (lane = neighbor column) -> next-layer B-fragment needs only an
// elementwise f32->f16 pack + one shfl_xor(16) half swap. Max via butterfly.
// ---------------------------------------------------------------------------
DEVINL v16h load_afrag(const uint4* __restrict__ frags, int fragid, int lane) {
    V16H a;
    const uint4* p = frags + (size_t)fragid * 64 + lane * 2;
    a.q[0] = p[0];
    a.q[1] = p[1];
    return a.v;
}

DEVINL v16h cvt16(const float* __restrict__ p) {   // 16 consecutive f32 -> v16h
    const float4* q = (const float4*)p;
    float4 a = q[0], b = q[1], c = q[2], d = q[3];
    V16H r;
    r.h[0] = (_Float16)a.x;  r.h[1] = (_Float16)a.y;  r.h[2] = (_Float16)a.z;  r.h[3] = (_Float16)a.w;
    r.h[4] = (_Float16)b.x;  r.h[5] = (_Float16)b.y;  r.h[6] = (_Float16)b.z;  r.h[7] = (_Float16)b.w;
    r.h[8] = (_Float16)c.x;  r.h[9] = (_Float16)c.y;  r.h[10] = (_Float16)c.z; r.h[11] = (_Float16)c.w;
    r.h[12] = (_Float16)d.x; r.h[13] = (_Float16)d.y; r.h[14] = (_Float16)d.z; r.h[15] = (_Float16)d.w;
    return r.v;
}

DEVINL void relu_pack(const v8f& c, const float* __restrict__ bias, int ot, int h, unsigned pk[4]) {
    const float4* pb = (const float4*)(bias + 16 * ot + 8 * h);
    float4 b0 = pb[0], b1 = pb[1];
    float r0 = fmaxf(c[0] + b0.x, 0.f), r1 = fmaxf(c[1] + b0.y, 0.f);
    float r2 = fmaxf(c[2] + b0.z, 0.f), r3 = fmaxf(c[3] + b0.w, 0.f);
    float r4 = fmaxf(c[4] + b1.x, 0.f), r5 = fmaxf(c[5] + b1.y, 0.f);
    float r6 = fmaxf(c[6] + b1.z, 0.f), r7 = fmaxf(c[7] + b1.w, 0.f);
    pk[0] = pk2(r0, r1); pk[1] = pk2(r2, r3); pk[2] = pk2(r4, r5); pk[3] = pk2(r6, r7);
}

DEVINL v16h mix_bfrag(const unsigned mine[4], const unsigned oth[4], int h) {
    V16H b;
    #pragma unroll
    for (int r = 0; r < 4; ++r) {
        unsigned rec = __shfl_xor(oth[r], 16, 32);
        b.u[r]     = h ? rec     : mine[r];
        b.u[4 + r] = h ? mine[r] : rec;
    }
    return b.v;
}

__global__ __launch_bounds__(256) void pointconv_kernel(
    const float* __restrict__ x, const float* __restrict__ pos,
    const int* __restrict__ nbr, const uint4* __restrict__ frags,
    const float* __restrict__ b1, const float* __restrict__ b2,
    const float* __restrict__ b3, float* __restrict__ out) {

    int lane = threadIdx.x & 31;
    int wave = threadIdx.x >> 5;
    int i    = blockIdx.x * 8 + wave;       // one point per wave
    int m    = lane & 15;
    int h    = lane >> 4;

    if (threadIdx.x < 36) __builtin_prefetch(((const char*)frags) + threadIdx.x * 1024, 0, 0);

    float pix = pos[3 * i], piy = pos[3 * i + 1], piz = pos[3 * i + 2];

    v8f dmax[8];
    #pragma unroll
    for (int t = 0; t < 8; ++t) dmax[t] = (v8f){0, 0, 0, 0, 0, 0, 0, 0};

    for (int ch = 0; ch < 4; ++ch) {                         // 4 chunks of 16 neighbors
        int j = nbr[(size_t)i * 64 + ch * 16 + m];
        const float* xj = x + (size_t)j * CIN;

        // Feature B-fragments: B element e <-> channel 32*kt + e + 16*h
        v16h bf0 = cvt16(xj + 16 * h);
        v16h bf1 = cvt16(xj + 32 + 16 * h);
        V16H bf2;
        #pragma unroll
        for (int r = 0; r < 8; ++r) bf2.u[r] = 0;
        if (h == 0) {
            bf2.h[0] = (_Float16)(pos[3 * j]     - pix);
            bf2.h[1] = (_Float16)(pos[3 * j + 1] - piy);
            bf2.h[2] = (_Float16)(pos[3 * j + 2] - piz);
        }

        // ---- Layer 1: 4 out-tiles x 3 k-tiles ----
        unsigned pk1[4][4];
        #pragma unroll
        for (int ot = 0; ot < 4; ++ot) {
            v8f c = {};
            c = __builtin_amdgcn_wmma_f32_16x16x32_f16(false, load_afrag(frags, ot * 3 + 0, lane), false, bf0,   (short)0, c, false, false);
            c = __builtin_amdgcn_wmma_f32_16x16x32_f16(false, load_afrag(frags, ot * 3 + 1, lane), false, bf1,   (short)0, c, false, false);
            c = __builtin_amdgcn_wmma_f32_16x16x32_f16(false, load_afrag(frags, ot * 3 + 2, lane), false, bf2.v, (short)0, c, false, false);
            relu_pack(c, b1, ot, h, pk1[ot]);
        }
        v16h bA[2];
        #pragma unroll
        for (int kt = 0; kt < 2; ++kt) {
            unsigned mine[4], oth[4];
            #pragma unroll
            for (int r = 0; r < 4; ++r) {
                mine[r] = h ? pk1[2 * kt + 1][r] : pk1[2 * kt][r];
                oth[r]  = h ? pk1[2 * kt][r]     : pk1[2 * kt + 1][r];
            }
            bA[kt] = mix_bfrag(mine, oth, h);
        }

        // ---- Layer 2: 4 out-tiles x 2 k-tiles ----
        unsigned pk2a[4][4];
        #pragma unroll
        for (int ot = 0; ot < 4; ++ot) {
            v8f c = {};
            c = __builtin_amdgcn_wmma_f32_16x16x32_f16(false, load_afrag(frags, 12 + ot * 2 + 0, lane), false, bA[0], (short)0, c, false, false);
            c = __builtin_amdgcn_wmma_f32_16x16x32_f16(false, load_afrag(frags, 12 + ot * 2 + 1, lane), false, bA[1], (short)0, c, false, false);
            relu_pack(c, b2, ot, h, pk2a[ot]);
        }
        v16h bB[2];
        #pragma unroll
        for (int kt = 0; kt < 2; ++kt) {
            unsigned mine[4], oth[4];
            #pragma unroll
            for (int r = 0; r < 4; ++r) {
                mine[r] = h ? pk2a[2 * kt + 1][r] : pk2a[2 * kt][r];
                oth[r]  = h ? pk2a[2 * kt][r]     : pk2a[2 * kt + 1][r];
            }
            bB[kt] = mix_bfrag(mine, oth, h);
        }

        // ---- Layer 3: 8 out-tiles x 2 k-tiles, bias+relu+running max ----
        #pragma unroll
        for (int ot = 0; ot < 8; ++ot) {
            v8f c = {};
            c = __builtin_amdgcn_wmma_f32_16x16x32_f16(false, load_afrag(frags, 20 + ot * 2 + 0, lane), false, bB[0], (short)0, c, false, false);
            c = __builtin_amdgcn_wmma_f32_16x16x32_f16(false, load_afrag(frags, 20 + ot * 2 + 1, lane), false, bB[1], (short)0, c, false, false);
            const float4* pb = (const float4*)(b3 + 16 * ot + 8 * h);
            float4 q0 = pb[0], q1 = pb[1];
            dmax[ot][0] = fmaxf(dmax[ot][0], fmaxf(c[0] + q0.x, 0.f));
            dmax[ot][1] = fmaxf(dmax[ot][1], fmaxf(c[1] + q0.y, 0.f));
            dmax[ot][2] = fmaxf(dmax[ot][2], fmaxf(c[2] + q0.z, 0.f));
            dmax[ot][3] = fmaxf(dmax[ot][3], fmaxf(c[3] + q0.w, 0.f));
            dmax[ot][4] = fmaxf(dmax[ot][4], fmaxf(c[4] + q1.x, 0.f));
            dmax[ot][5] = fmaxf(dmax[ot][5], fmaxf(c[5] + q1.y, 0.f));
            dmax[ot][6] = fmaxf(dmax[ot][6], fmaxf(c[6] + q1.z, 0.f));
            dmax[ot][7] = fmaxf(dmax[ot][7], fmaxf(c[7] + q1.w, 0.f));
        }
    }

    // Butterfly max over the 16 neighbor columns within each half-group.
    #pragma unroll
    for (int mask = 1; mask <= 8; mask <<= 1) {
        #pragma unroll
        for (int t = 0; t < 8; ++t) {
            #pragma unroll
            for (int r = 0; r < 8; ++r) {
                float o = __shfl_xor(dmax[t][r], mask, 32);
                dmax[t][r] = fmaxf(dmax[t][r], o);
            }
        }
    }

    if (m == 0) {   // lanes 0 and 16 hold the 128 channel maxima (64 each)
        float* op = out + (size_t)i * 128;
        #pragma unroll
        for (int t = 0; t < 8; ++t) {
            float4 v0 = make_float4(dmax[t][0], dmax[t][1], dmax[t][2], dmax[t][3]);
            float4 v1 = make_float4(dmax[t][4], dmax[t][5], dmax[t][6], dmax[t][7]);
            float4* dst = (float4*)(op + 16 * t + 8 * h);
            dst[0] = v0;
            dst[1] = v1;
        }
    }
}

// ---------------------------------------------------------------------------
// Kernel 3: farthest point sampling, one 1024-thread block per graph.
// ---------------------------------------------------------------------------
__global__ __launch_bounds__(1024) void fps_kernel(const float* __restrict__ pos,
                                                   int* __restrict__ fps) {
    __shared__ float sx[NPER], sy[NPER], sz[NPER];
    __shared__ float rbv[32];
    __shared__ int rbi[32];
    __shared__ int s_sel;

    int g = blockIdx.x;
    int base = g * NPER;
    int tid = threadIdx.x;

    float px[4], py[4], pz[4], dist[4];
    #pragma unroll
    for (int q = 0; q < 4; ++q) {
        int t = tid + q * 1024;
        float a = pos[3 * (base + t)], b = pos[3 * (base + t) + 1], c = pos[3 * (base + t) + 2];
        sx[t] = a; sy[t] = b; sz[t] = c;
        px[q] = a; py[q] = b; pz[q] = c;
        dist[q] = __builtin_inff();
    }
    if (tid == 0) { fps[g * MFPS] = base; s_sel = 0; }
    __syncthreads();

    int sel = 0;
    for (int s = 1; s < MFPS; ++s) {
        float cx = sx[sel], cy = sy[sel], cz = sz[sel];
        float bv = -1.0f; int bi = 0x7FFFFFFF;
        #pragma unroll
        for (int q = 0; q < 4; ++q) {
            float dx = px[q] - cx, dy = py[q] - cy, dz = pz[q] - cz;
            float d = dx * dx + dy * dy + dz * dz;
            dist[q] = fminf(dist[q], d);
            int t = tid + q * 1024;
            if (dist[q] > bv || (dist[q] == bv && t < bi)) { bv = dist[q]; bi = t; }
        }
        #pragma unroll
        for (int off = 16; off; off >>= 1) {
            float v2 = __shfl_down(bv, off, 32);
            int   i2 = __shfl_down(bi, off, 32);
            if (v2 > bv || (v2 == bv && i2 < bi)) { bv = v2; bi = i2; }
        }
        if ((tid & 31) == 0) { rbv[tid >> 5] = bv; rbi[tid >> 5] = bi; }
        __syncthreads();
        if (tid < 32) {
            bv = rbv[tid]; bi = rbi[tid];
            #pragma unroll
            for (int off = 16; off; off >>= 1) {
                float v2 = __shfl_down(bv, off, 32);
                int   i2 = __shfl_down(bi, off, 32);
                if (v2 > bv || (v2 == bv && i2 < bi)) { bv = v2; bi = i2; }
            }
            if (tid == 0) { s_sel = bi; fps[g * MFPS + s] = base + bi; }
        }
        __syncthreads();
        sel = s_sel;
    }
}

// ---------------------------------------------------------------------------
// Kernel 4: gather sampled outputs / positions / batch ids.
// ---------------------------------------------------------------------------
__global__ __launch_bounds__(128) void gather_kernel(const float* __restrict__ pos,
                                                     const float* __restrict__ pout,
                                                     const int* __restrict__ fps,
                                                     float* __restrict__ xo,
                                                     float* __restrict__ po,
                                                     int* __restrict__ bo) {
    int s = blockIdx.x;
    int c = threadIdx.x;
    int src = fps[s];
    xo[(size_t)s * 128 + c] = pout[(size_t)src * 128 + c];
    if (c < 3) po[s * 3 + c] = pos[src * 3 + c];
    if (c == 0) bo[s] = s >> 10;   // s / MFPS
}

// ---------------------------------------------------------------------------
extern "C" void kernel_launch(void* const* d_in, const int* in_sizes, int n_in,
                              void* d_out, int out_size, void* d_ws, size_t ws_size,
                              hipStream_t stream) {
    const float* x   = (const float*)d_in[0];
    const float* pos = (const float*)d_in[1];
    // d_in[2] = batch (unused; batch ids are recomputed)
    const float* W1 = (const float*)d_in[3];
    const float* b1 = (const float*)d_in[4];
    const float* W2 = (const float*)d_in[5];
    const float* b2 = (const float*)d_in[6];
    const float* W3 = (const float*)d_in[7];
    const float* b3 = (const float*)d_in[8];

    char* ws = (char*)d_ws;
    int*       nbr   = (int*)(ws + WS_NBR);
    _Float16*  fragh = (_Float16*)(ws + WS_FRAG);
    float*     pout  = (float*)(ws + WS_OUT);
    int*       fps   = (int*)(ws + WS_FPS);

    float* xo = (float*)d_out;
    float* po = xo + (size_t)BG * MFPS * 128;
    int*   bo = (int*)(po + (size_t)BG * MFPS * 3);

    prep_frags_kernel<<<(36 * 512 + 255) / 256, 256, 0, stream>>>(W1, W2, W3, fragh);
    radius_kernel<<<NPTS, 256, 0, stream>>>(pos, nbr);
    fps_kernel<<<BG, 1024, 0, stream>>>(pos, fps);
    pointconv_kernel<<<NPTS / 8, 256, 0, stream>>>(x, pos, nbr, (const uint4*)fragh,
                                                   b1, b2, b3, pout);
    gather_kernel<<<BG * MFPS, 128, 0, stream>>>(pos, pout, fps, xo, po, bo);
}
